// LSTMDecoder_53051436040585
// MI455X (gfx1250) — matrix operand order repaired and checked by reading, not compiled, not run on hardware
//
#include <hip/hip_runtime.h>
#include <hip/hip_bf16.h>
#include <stdint.h>

// Problem dims (match reference)
#define B_   64
#define T_   128
#define S_   1024
#define I_   512
#define H_   512
#define G4   2048   // 4*H
#define KCAT 1536   // (I + H) from W_ih  +  H from W_hh

typedef __attribute__((ext_vector_type(16))) __bf16 v16bf;
typedef __attribute__((ext_vector_type(8)))  __bf16 v8bf;
typedef __attribute__((ext_vector_type(8)))  float  v8f;

static __device__ __forceinline__ unsigned short f2bf(float f) {
  union { float f; unsigned int u; } cv; cv.f = f;
  unsigned int u = cv.u;
  u += 0x7fffu + ((u >> 16) & 1u);      // round-to-nearest-even
  return (unsigned short)(u >> 16);
}
static __device__ __forceinline__ float bflo(unsigned int v) {
  union { unsigned int u; float f; } cv; cv.u = v << 16;
  return cv.f;
}
static __device__ __forceinline__ float bfhi(unsigned int v) {
  union { unsigned int u; float f; } cv; cv.u = v & 0xffff0000u;
  return cv.f;
}
static __device__ __forceinline__ float sigmoidf_(float x) {
  return 1.0f / (1.0f + __expf(-x));
}

// Load a 16-element bf16 WMMA fragment as two contiguous B128 loads.
static __device__ __forceinline__ v16bf ld_frag(const unsigned short* p0,
                                                const unsigned short* p1) {
  v8bf lo = *(const v8bf*)p0;
  v8bf hi = *(const v8bf*)p1;
  v16bf r;
#pragma unroll
  for (int i = 0; i < 8; ++i) { r[i] = lo[i]; r[i + 8] = hi[i]; }
  return r;
}

// ---------------------------------------------------------------------------
// One-time prep: encoder_hy [S,B,H] f32 -> ebf [B,S,H] bf16 (64 MB, L2-resident)
// ---------------------------------------------------------------------------
__global__ void k_prep_enhy(const float* __restrict__ enhy,
                            unsigned short* __restrict__ ebf) {
  long i = (long)blockIdx.x * blockDim.x + threadIdx.x;   // over B*S*H
  int h  = (int)(i % H_);
  long bs = i / H_;
  int s  = (int)(bs % S_);
  int b  = (int)(bs / S_);
  ebf[i] = f2bf(enhy[((long)s * B_ + b) * H_ + h]);
}

// ---------------------------------------------------------------------------
// One-time prep: W_cat bf16 [2048][1536] = [W_ih | W_hh], bias = b_ih + b_hh
// ---------------------------------------------------------------------------
__global__ void k_prep_w(const float* __restrict__ Wih, const float* __restrict__ Whh,
                         const float* __restrict__ bih, const float* __restrict__ bhh,
                         unsigned short* __restrict__ Wbf, float* __restrict__ bias) {
  long i = (long)blockIdx.x * blockDim.x + threadIdx.x;   // over G4*KCAT
  int k = (int)(i % KCAT);
  int g = (int)(i / KCAT);
  float v = (k < (I_ + H_)) ? Wih[(long)g * (I_ + H_) + k]
                            : Whh[(long)g * H_ + (k - (I_ + H_))];
  Wbf[i] = f2bf(v);
  if (i < G4) bias[i] = bih[i] + bhh[i];
}

// ---------------------------------------------------------------------------
// One-time prep: state init
// ---------------------------------------------------------------------------
__global__ void k_init_state(const float* __restrict__ h0, const float* __restrict__ c0,
                             float* __restrict__ h_cur, float* __restrict__ c_cur) {
  int i = blockIdx.x * blockDim.x + threadIdx.x;
  h_cur[i] = h0[i];
  c_cur[i] = c0[i];
}

// ---------------------------------------------------------------------------
// Per-step: fused attention (scores -> softmax -> tanh(context)) + pack the
// bf16 activation row A[b] = [ bf16(x[b,t,:]) | bf16(h_attn) | bf16(h_prev) ].
// One workgroup per batch element; scores + softmax live in LDS.
// Bandwidth-critical: streams the L2-resident bf16 encoder copy twice.
// ---------------------------------------------------------------------------
__global__ void __launch_bounds__(256)
k_attention(const unsigned short* __restrict__ ebf,  // [B][S][H] bf16
            const float* __restrict__ h_cur,         // [B][H]
            const float* __restrict__ x,             // [B][T][I]
            int t,
            unsigned short* __restrict__ A,          // [B][KCAT] bf16
            float* __restrict__ attn_out)            // [B][S] (d_out attn region)
{
  __shared__ float sc[S_];     // 4 KB scores / attn weights
  __shared__ float hv[H_];     // 2 KB hidden vector
  __shared__ float red[256];   // 1 KB reduction scratch
  const int b = blockIdx.x, tid = threadIdx.x;
  const unsigned short* Eb = ebf + (long)b * S_ * H_;

  for (int k = tid; k < H_; k += 256) hv[k] = h_cur[b * H_ + k];
  __syncthreads();

  // Pass 1: scores[s] = <E[b,s,:], h>; rows contiguous -> B128 loads (8 bf16 each)
  for (int s = tid; s < S_; s += 256) {
    const uint4* row = (const uint4*)(Eb + (long)s * H_);
    float acc = 0.f;
#pragma unroll 4
    for (int k = 0; k < H_ / 8; ++k) {
      uint4 v = row[k];
      const float* hp = &hv[8 * k];
      acc += bflo(v.x) * hp[0] + bfhi(v.x) * hp[1];
      acc += bflo(v.y) * hp[2] + bfhi(v.y) * hp[3];
      acc += bflo(v.z) * hp[4] + bfhi(v.z) * hp[5];
      acc += bflo(v.w) * hp[6] + bfhi(v.w) * hp[7];
    }
    sc[s] = acc;
  }
  __syncthreads();

  // Softmax over S: max reduce
  float m = -3.4e38f;
  for (int s = tid; s < S_; s += 256) m = fmaxf(m, sc[s]);
  red[tid] = m; __syncthreads();
  for (int off = 128; off > 0; off >>= 1) {
    if (tid < off) red[tid] = fmaxf(red[tid], red[tid + off]);
    __syncthreads();
  }
  m = red[0]; __syncthreads();

  // exp + sum reduce
  float ssum = 0.f;
  for (int s = tid; s < S_; s += 256) { float e = __expf(sc[s] - m); sc[s] = e; ssum += e; }
  red[tid] = ssum; __syncthreads();
  for (int off = 128; off > 0; off >>= 1) {
    if (tid < off) red[tid] += red[tid + off];
    __syncthreads();
  }
  const float inv = 1.0f / red[0]; __syncthreads();

  for (int s = tid; s < S_; s += 256) {
    float a = sc[s] * inv;
    sc[s] = a;
    attn_out[b * S_ + s] = a;   // last timestep's write survives => attns[-1]
  }
  __syncthreads();

  // Pass 2: context[2h,2h+1] = sum_s attn[s]*E[b,s,2h..2h+1]; one bf16-pair
  // (b32 load) per thread per s; a wave's 32 lanes cover 128 consecutive bytes.
  unsigned short* Ab = A + (long)b * KCAT;
  for (int hp = tid; hp < H_ / 2; hp += 256) {
    float c0 = 0.f, c1 = 0.f;
    const unsigned int* col = (const unsigned int*)(Eb + 2 * hp);
#pragma unroll 4
    for (int s = 0; s < S_; ++s) {
      unsigned int v = col[(long)s * (H_ / 2)];
      float a = sc[s];
      c0 += a * bflo(v);
      c1 += a * bfhi(v);
    }
    Ab[I_ + 2 * hp]     = f2bf(tanhf(c0));
    Ab[I_ + 2 * hp + 1] = f2bf(tanhf(c1));
  }
  const float* xr = x + ((long)b * T_ + t) * I_;
  for (int k = tid; k < I_; k += 256) Ab[k] = f2bf(xr[k]);
  for (int h = tid; h < H_; h += 256) Ab[I_ + H_ + h] = f2bf(hv[h]);
}

// ---------------------------------------------------------------------------
// Per-step: gate GEMM  gates[64,2048] = A[64,1536] x Wbf^T + bias
// via v_wmma_f32_16x16x32_bf16, software-pipelined: next K-chunk's fragments
// are prefetched into a second register set while the current chunk's four
// WMMAs issue, so waits are not loadcnt==0 full stalls.
// Fragment layouts follow CDNA5 ISA §7.12.2 (wave32).
// ---------------------------------------------------------------------------
__global__ void __launch_bounds__(256)
k_gemm_gates(const unsigned short* __restrict__ A,    // [64][1536] bf16
             const unsigned short* __restrict__ Wbf,  // [2048][1536] bf16
             const float* __restrict__ bias,          // [2048]
             float* __restrict__ gates)               // [64][2048]
{
  const int lane  = threadIdx.x & 31;
  const int wid   = blockIdx.x * (blockDim.x >> 5) + (threadIdx.x >> 5); // 0..127
  const int g0    = wid * 16;
  const int n     = lane & 15;
  const int khalf = lane >> 4;            // 0: low-K half, 1: high-K half

  // Per-lane base pointers for fragment loads.
  // B fragment (32x16 K x N): lane holds column n; lane<16: K {0..7,16..23},
  //   lane>=16: K {8..15,24..31}; element (k,n) = W[g0+n][k0+k]   (gates = A x W^T)
  const unsigned short* Wrow = Wbf + (long)(g0 + n) * KCAT + khalf * 8;
  // A fragment (16x32): lane row = m = lane&15; lane<16: K {0..7,16..23},
  //   lane>=16: K {8..15,24..31}
  const unsigned short* Ar0 = A + (long)( 0 + n) * KCAT + khalf * 8;
  const unsigned short* Ar1 = A + (long)(16 + n) * KCAT + khalf * 8;
  const unsigned short* Ar2 = A + (long)(32 + n) * KCAT + khalf * 8;
  const unsigned short* Ar3 = A + (long)(48 + n) * KCAT + khalf * 8;

  const float bval = bias[g0 + n];
  v8f acc0, acc1, acc2, acc3;
#pragma unroll
  for (int r = 0; r < 8; ++r) { acc0[r] = bval; acc1[r] = bval; acc2[r] = bval; acc3[r] = bval; }

  // Prologue: load K-chunk 0
  v16bf bc  = ld_frag(Wrow, Wrow + 16);
  v16bf a0c = ld_frag(Ar0, Ar0 + 16);
  v16bf a1c = ld_frag(Ar1, Ar1 + 16);
  v16bf a2c = ld_frag(Ar2, Ar2 + 16);
  v16bf a3c = ld_frag(Ar3, Ar3 + 16);

  for (int k0 = 0; k0 < KCAT; k0 += 32) {
    int kn = k0 + 32;
    if (kn >= KCAT) kn = 0;               // tail prefetch wraps (discarded)
    // Prefetch next chunk into the alternate register set.
    v16bf bn  = ld_frag(Wrow + kn, Wrow + kn + 16);
    v16bf a0n = ld_frag(Ar0 + kn, Ar0 + kn + 16);
    v16bf a1n = ld_frag(Ar1 + kn, Ar1 + kn + 16);
    v16bf a2n = ld_frag(Ar2 + kn, Ar2 + kn + 16);
    v16bf a3n = ld_frag(Ar3 + kn, Ar3 + kn + 16);

    // Consume current chunk: one B fragment reused across 4 M-tiles.
    acc0 = __builtin_amdgcn_wmma_f32_16x16x32_bf16(false, a0c, false, bc, (short)0, acc0, false, false);
    acc1 = __builtin_amdgcn_wmma_f32_16x16x32_bf16(false, a1c, false, bc, (short)0, acc1, false, false);
    acc2 = __builtin_amdgcn_wmma_f32_16x16x32_bf16(false, a2c, false, bc, (short)0, acc2, false, false);
    acc3 = __builtin_amdgcn_wmma_f32_16x16x32_bf16(false, a3c, false, bc, (short)0, acc3, false, false);

    bc = bn; a0c = a0n; a1c = a1n; a2c = a2n; a3c = a3n;
  }

  // C/D layout: VGPR r -> row (r + khalf*8), col n
#pragma unroll
  for (int r = 0; r < 8; ++r) {
    const int mrow = r + khalf * 8;
    gates[(long)( 0 + mrow) * G4 + g0 + n] = acc0[r];
    gates[(long)(16 + mrow) * G4 + g0 + n] = acc1[r];
    gates[(long)(32 + mrow) * G4 + g0 + n] = acc2[r];
    gates[(long)(48 + mrow) * G4 + g0 + n] = acc3[r];
  }
}

// ---------------------------------------------------------------------------
// Per-step: LSTM cell elementwise, writes seq output + running h/c (+ final
// h/c regions of d_out every step; last step's values persist).
// ---------------------------------------------------------------------------
__global__ void k_cell(const float* __restrict__ gates,
                       float* __restrict__ h_cur, float* __restrict__ c_cur,
                       float* __restrict__ out_seq,
                       float* __restrict__ out_h, float* __restrict__ out_c, int t)
{
  const int i = blockIdx.x * blockDim.x + threadIdx.x;  // over B*H
  const int b = i / H_, h = i % H_;
  const float* gr = gates + (long)b * G4;
  const float gi = gr[h];
  const float gf = gr[H_ + h];
  const float gg = gr[2 * H_ + h];
  const float go = gr[3 * H_ + h];
  const float cn = sigmoidf_(gf) * c_cur[i] + sigmoidf_(gi) * tanhf(gg);
  const float hn = sigmoidf_(go) * tanhf(cn);
  c_cur[i] = cn;
  h_cur[i] = hn;
  out_seq[((long)b * T_ + t) * H_ + h] = hn;
  out_h[i] = hn;
  out_c[i] = cn;
}

// ---------------------------------------------------------------------------
extern "C" void kernel_launch(void* const* d_in, const int* in_sizes, int n_in,
                              void* d_out, int out_size, void* d_ws, size_t ws_size,
                              hipStream_t stream) {
  const float* x    = (const float*)d_in[0];  // [B,T,I]
  const float* h0   = (const float*)d_in[1];  // [B,H]
  const float* c0   = (const float*)d_in[2];  // [B,H]
  const float* enhy = (const float*)d_in[3];  // [S,B,H]
  const float* Wih  = (const float*)d_in[4];  // [4H, I+H]
  const float* Whh  = (const float*)d_in[5];  // [4H, H]
  const float* bih  = (const float*)d_in[6];  // [4H]
  const float* bhh  = (const float*)d_in[7];  // [4H]

  float* out       = (float*)d_out;
  float* out_seq   = out;                          // B*T*H
  float* out_h     = out_seq + (long)B_ * T_ * H_; // B*H
  float* out_c     = out_h + B_ * H_;              // B*H
  float* out_attn  = out_c + B_ * H_;              // B*S

  char* ws = (char*)d_ws;
  unsigned short* ebf  = (unsigned short*)ws; ws += (long)B_ * S_ * H_ * 2;  // 64 MB
  unsigned short* Wbf  = (unsigned short*)ws; ws += (long)G4 * KCAT * 2;     // 6 MB
  float*          bias = (float*)ws;          ws += (long)G4 * 4;
  unsigned short* Abuf = (unsigned short*)ws; ws += (long)B_ * KCAT * 2;
  float*          gts  = (float*)ws;          ws += (long)B_ * G4 * 4;
  float*          h_cur= (float*)ws;          ws += (long)B_ * H_ * 4;
  float*          c_cur= (float*)ws;

  // One-time preparation (inside capture: deterministic, re-run every replay)
  k_prep_enhy <<<(B_ * S_ * H_) / 256, 256, 0, stream>>>(enhy, ebf);
  k_prep_w    <<<(G4 * KCAT) / 256,    256, 0, stream>>>(Wih, Whh, bih, bhh, Wbf, bias);
  k_init_state<<<(B_ * H_) / 256,      256, 0, stream>>>(h0, c0, h_cur, c_cur);

  for (int t = 0; t < T_; ++t) {
    k_attention <<<B_,  256, 0, stream>>>(ebf, h_cur, x, t, Abuf, out_attn);
    k_gemm_gates<<<16,  256, 0, stream>>>(Abuf, Wbf, bias, gts);
    k_cell      <<<(B_ * H_) / 256, 256, 0, stream>>>(gts, h_cur, c_cur,
                                                      out_seq, out_h, out_c, t);
  }
}
